// AttentionBlock_44762149159315
// MI455X (gfx1250) — compile-verified, matrix operand order checked by reading, not run on hardware
//
#include <hip/hip_runtime.h>
#include <hip/hip_bf16.h>

// ---------------------------------------------------------------------------
// AttentionBlock for MI455X (gfx1250, wave32, WMMA 16x16x32 f16 -> f32 acc)
//
// B=8, C=512, N=H*W=1024, HEADS=4, D=128.  ~34.4 GFLOP, ~42 MB working set
// (fully L2-resident on 192MB L2) -> compute/latency bound, so: all matmuls
// on v_wmma_f32_16x16x32_f16, k-loops software-pipelined (double-buffered
// fragments) so WMMAs overlap the next tile's global_load_b128s.
// ---------------------------------------------------------------------------

typedef _Float16 v16h __attribute__((ext_vector_type(16)));
typedef _Float16 v8h  __attribute__((ext_vector_type(8)));
typedef float    v8f  __attribute__((ext_vector_type(8)));
typedef float    v4f  __attribute__((ext_vector_type(4)));

#define NB   8
#define CCH  512
#define NTOK 1024
#define NH   4
#define DK   128

// One 16-f16 WMMA fragment: per-lane two contiguous 8-f16 (16B) chunks at
// +0 and +16 elements (K split per ISA 16-bit A/B fragment layout).
__device__ __forceinline__ v16h load_frag(const _Float16* p) {
  union { v16h v; v8h h[2]; } u;
  u.h[0] = *(const v8h*)(p);
  u.h[1] = *(const v8h*)(p + 16);
  return u.v;
}

__device__ __forceinline__ v8f wmma16(v16h a, v16h b, v8f c) {
  // (neg_a, A, neg_b, B, c_mod, C, reuse_a, reuse_b)
  return __builtin_amdgcn_wmma_f32_16x16x32_f16(false, a, false, b,
                                                (short)0, c, false, false);
}

// ---------------------------------------------------------------------------
// Kernel 0: batched transpose + f32->f16 cast.  dst[b][c][r] = src[b][r][c]
// ---------------------------------------------------------------------------
__global__ void __launch_bounds__(256)
transpose_cast(const float* __restrict__ src, _Float16* __restrict__ dst,
               int rows, int cols) {
  __shared__ float tile[32][33];
  const size_t boff = (size_t)blockIdx.z * rows * cols;
  const int cx = blockIdx.x * 32 + threadIdx.x;   // col
  const int ry = blockIdx.y * 32 + threadIdx.y;   // row
#pragma unroll
  for (int j = 0; j < 32; j += 8)
    tile[threadIdx.y + j][threadIdx.x] = src[boff + (size_t)(ry + j) * cols + cx];
  __syncthreads();
  const int rx = blockIdx.y * 32 + threadIdx.x;   // now row index
  const int cy = blockIdx.x * 32 + threadIdx.y;   // now col index
#pragma unroll
  for (int j = 0; j < 32; j += 8)
    dst[boff + (size_t)(cy + j) * rows + rx] = (_Float16)tile[threadIdx.x][threadIdx.y + j];
}

// ---------------------------------------------------------------------------
// Kernel 1: QKV projection.  One wave -> 32 rows x 64 cols tile (2 A-row
// fragments x 4 B fragments = 8 WMMAs per k-step, double-buffered loads).
// Scatters into q/k [B,H,N,128] and vT [B,H,128,N].
// ---------------------------------------------------------------------------
__global__ void __launch_bounds__(32)
qkv_gemm(const _Float16* __restrict__ xh,   // [B][N][C] f16
         const _Float16* __restrict__ WpT,  // [1536][512] f16 (Wp transposed)
         const float* __restrict__ bp,      // [1536]
         _Float16* __restrict__ qh,         // [B][H][N][128]
         _Float16* __restrict__ kh,         // [B][H][N][128]
         _Float16* __restrict__ vT) {       // [B][H][128][N]
  const int lane  = threadIdx.x;
  const int hf    = lane >> 4;
  const int l16   = lane & 15;
  const int kbase = hf * 8;
  const int roff  = hf * 8;
  const int n0 = blockIdx.x * 32;
  const int j0 = blockIdx.y * 64;
  const int b  = blockIdx.z;

  const _Float16* A0 = xh + ((size_t)b * NTOK + (n0 + l16)) * CCH + kbase;
  const _Float16* A1 = A0 + (size_t)16 * CCH;
  const _Float16* Bp[4];
#pragma unroll
  for (int t = 0; t < 4; ++t)
    Bp[t] = WpT + (size_t)(j0 + t * 16 + l16) * CCH + kbase;

  // ---- software-pipelined k-loop (double-buffered fragments) ----
  v16h a0c = load_frag(A0), a1c = load_frag(A1);
  v16h bc[4];
#pragma unroll
  for (int t = 0; t < 4; ++t) bc[t] = load_frag(Bp[t]);
  v8f acc0[4] = {}, acc1[4] = {};
#pragma unroll
  for (int k0 = 0; k0 < CCH; k0 += 32) {
    v16h a0n = a0c, a1n = a1c, bn[4] = {bc[0], bc[1], bc[2], bc[3]};
    const int kn = k0 + 32;
    if (kn < CCH) {
      a0n = load_frag(A0 + kn);
      a1n = load_frag(A1 + kn);
#pragma unroll
      for (int t = 0; t < 4; ++t) bn[t] = load_frag(Bp[t] + kn);
    }
#pragma unroll
    for (int t = 0; t < 4; ++t) {
      acc0[t] = wmma16(a0c, bc[t], acc0[t]);
      acc1[t] = wmma16(a1c, bc[t], acc1[t]);
    }
    a0c = a0n; a1c = a1n;
#pragma unroll
    for (int t = 0; t < 4; ++t) bc[t] = bn[t];
  }

  // ---- epilogue: bias + scatter into q / k / vT ----
#pragma unroll
  for (int t = 0; t < 4; ++t) {
    const int j0t  = j0 + t * 16;          // 16-col tile never crosses a
    const int head = j0t / 384;            // q/k/v segment boundary
    const int r0   = j0t % 384;
    const int seg  = r0 / 128;             // 0=q 1=k 2=v
    const int d    = (r0 % 128) + l16;
    const float bias = bp[j0t + l16];
    if (seg == 2) {
      _Float16* dst = vT + (((size_t)b * NH + head) * DK + d) * NTOK + n0 + roff;
#pragma unroll
      for (int g = 0; g < 8; ++g) {
        dst[g]      = (_Float16)(acc0[t][g] + bias);
        dst[g + 16] = (_Float16)(acc1[t][g] + bias);
      }
    } else {
      _Float16* base = (seg == 0) ? qh : kh;
      _Float16* p0 = base + (((size_t)b * NH + head) * NTOK + n0 + roff) * DK + d;
#pragma unroll
      for (int g = 0; g < 8; ++g) {
        p0[(size_t)g * DK]        = (_Float16)(acc0[t][g] + bias);
        p0[(size_t)(g + 16) * DK] = (_Float16)(acc1[t][g] + bias);
      }
    }
  }
}

// ---------------------------------------------------------------------------
// Kernel 2: flash attention.  One wave owns 16 query rows of one (b,head).
// Online softmax over 32-key blocks: 8 QK^T WMMAs + 8 PV WMMAs per block.
// First half of the V fragments is loaded BEFORE the softmax so the
// exp/shuffle VALU chain hides the global load latency; next K tile is
// prefetched at iteration start.  P re-layout (C->A frag) via per-wave LDS.
// ---------------------------------------------------------------------------
__global__ void __launch_bounds__(32)
attention_fa(const _Float16* __restrict__ qh, const _Float16* __restrict__ kh,
             const _Float16* __restrict__ vT, _Float16* __restrict__ attn) {
  __shared__ alignas(16) _Float16 lds_p[16][40];
  const int lane  = threadIdx.x;
  const int hf    = lane >> 4;
  const int l16   = lane & 15;
  const int kbase = hf * 8;
  const int roff  = hf * 8;
  const int n0 = blockIdx.x * 16;
  const int h  = blockIdx.y;
  const int b  = blockIdx.z;
  const float scale = 0.08838834764831845f;  // 1/sqrt(128)

  // Q fragments: 16 rows x 128 d -> 4 A-fragments of 16x32 (held in regs)
  const _Float16* Qb = qh + (((size_t)b * NH + h) * NTOK + n0 + l16) * DK;
  v16h qf[4];
#pragma unroll
  for (int kk = 0; kk < 4; ++kk) qf[kk] = load_frag(Qb + kk * 32 + kbase);

  v8f o[8] = {};                // O accumulators: 16 x 128 f32
  float mi[8], li[8];
#pragma unroll
  for (int g = 0; g < 8; ++g) { mi[g] = -1e30f; li[g] = 0.0f; }

  const _Float16* Kb = kh + ((size_t)b * NH + h) * NTOK * DK;
  const _Float16* Vb = vT + ((size_t)b * NH + h) * DK * NTOK;

  for (int j0 = 0; j0 < NTOK; j0 += 32) {
    if (j0 + 32 < NTOK) {   // prefetch next key tile
      __builtin_prefetch(Kb + (size_t)(j0 + 32 + l16) * DK + kbase, 0, 3);
      __builtin_prefetch(Kb + (size_t)(j0 + 48 + l16) * DK + kbase, 0, 3);
    }
    // ---- S = Q K^T (16 rows x 32 keys), f32 accum ----
    v8f s0 = {}, s1 = {};
    const _Float16* K0 = Kb + (size_t)(j0 + l16) * DK;        // keys j0..+15
    const _Float16* K1 = K0 + (size_t)16 * DK;                // keys +16..+31
#pragma unroll
    for (int kk = 0; kk < 4; ++kk) {
      s0 = wmma16(qf[kk], load_frag(K0 + kk * 32 + kbase), s0);
      s1 = wmma16(qf[kk], load_frag(K1 + kk * 32 + kbase), s1);
    }
    // ---- start first-half V loads; they drain during the softmax ----
    v16h vf[4];
#pragma unroll
    for (int t = 0; t < 4; ++t)
      vf[t] = load_frag(Vb + (size_t)(t * 16 + l16) * NTOK + j0 + kbase);

    // ---- online softmax (row stats via 16-lane half-wave reductions) ----
#pragma unroll
    for (int g = 0; g < 8; ++g) {
      const float a0 = s0[g] * scale, a1 = s1[g] * scale;
      float m = fmaxf(a0, a1);
      m = fmaxf(m, __shfl_xor(m, 1));
      m = fmaxf(m, __shfl_xor(m, 2));
      m = fmaxf(m, __shfl_xor(m, 4));
      m = fmaxf(m, __shfl_xor(m, 8));
      const float mn    = fmaxf(mi[g], m);
      const float alpha = __expf(mi[g] - mn);
      const float p0 = __expf(a0 - mn);
      const float p1 = __expf(a1 - mn);
      s0[g] = p0; s1[g] = p1;
      float rs = p0 + p1;
      rs += __shfl_xor(rs, 1);
      rs += __shfl_xor(rs, 2);
      rs += __shfl_xor(rs, 4);
      rs += __shfl_xor(rs, 8);
      li[g] = li[g] * alpha + rs;
      mi[g] = mn;
#pragma unroll
      for (int t = 0; t < 8; ++t) o[t][g] *= alpha;
    }
    // ---- C-layout -> A-layout for P through LDS ----
#pragma unroll
    for (int g = 0; g < 8; ++g) {
      lds_p[roff + g][l16]      = (_Float16)s0[g];
      lds_p[roff + g][16 + l16] = (_Float16)s1[g];
    }
    __syncthreads();                       // 1-wave group: orders DScnt
    const v16h pf = load_frag(&lds_p[l16][kbase]);
    // ---- O += P @ V  (32-key reduction = one WMMA per 16-d tile) ----
#pragma unroll
    for (int t = 0; t < 4; ++t) o[t] = wmma16(pf, vf[t], o[t]);
#pragma unroll
    for (int t = 4; t < 8; ++t) {
      const _Float16* Vp = Vb + (size_t)(t * 16 + l16) * NTOK + j0 + kbase;
      o[t] = wmma16(pf, load_frag(Vp), o[t]);
    }
    __syncthreads();                       // before lds_p is overwritten
  }
  // ---- finalize: O / l -> attn[B][N][512] f16 ----
  _Float16* Ob = attn + ((size_t)b * NTOK + n0 + roff) * (NH * DK) + h * DK + l16;
#pragma unroll
  for (int t = 0; t < 8; ++t)
#pragma unroll
    for (int g = 0; g < 8; ++g)
      Ob[(size_t)g * (NH * DK) + t * 16] = (_Float16)(o[t][g] / li[g]);
}

// ---------------------------------------------------------------------------
// Kernel 3: output projection + bias + residual, write [B][C][N] f32.
// One wave -> 32 rows x 64 cols, double-buffered like kernel 1.  D-fragment
// rows are consecutive n for a fixed lane/column -> float4 loads/stores.
// ---------------------------------------------------------------------------
__global__ void __launch_bounds__(32)
out_proj(const _Float16* __restrict__ attn,  // [B][N][512]
         const _Float16* __restrict__ WoT,   // [512(c)][512(k)]
         const float* __restrict__ bo,       // [512]
         const float* __restrict__ x,        // [B][C][N] f32 (residual)
         float* __restrict__ out) {          // [B][C][N] f32
  const int lane  = threadIdx.x;
  const int hf    = lane >> 4;
  const int l16   = lane & 15;
  const int kbase = hf * 8;
  const int roff  = hf * 8;
  const int n0 = blockIdx.x * 32;
  const int c0 = blockIdx.y * 64;
  const int b  = blockIdx.z;

  const _Float16* A0 = attn + ((size_t)b * NTOK + n0 + l16) * CCH + kbase;
  const _Float16* A1 = A0 + (size_t)16 * CCH;
  const _Float16* Bp[4];
#pragma unroll
  for (int t = 0; t < 4; ++t)
    Bp[t] = WoT + (size_t)(c0 + t * 16 + l16) * CCH + kbase;

  v16h a0c = load_frag(A0), a1c = load_frag(A1);
  v16h bc[4];
#pragma unroll
  for (int t = 0; t < 4; ++t) bc[t] = load_frag(Bp[t]);
  v8f acc0[4] = {}, acc1[4] = {};
#pragma unroll
  for (int k0 = 0; k0 < CCH; k0 += 32) {
    v16h a0n = a0c, a1n = a1c, bn[4] = {bc[0], bc[1], bc[2], bc[3]};
    const int kn = k0 + 32;
    if (kn < CCH) {
      a0n = load_frag(A0 + kn);
      a1n = load_frag(A1 + kn);
#pragma unroll
      for (int t = 0; t < 4; ++t) bn[t] = load_frag(Bp[t] + kn);
    }
#pragma unroll
    for (int t = 0; t < 4; ++t) {
      acc0[t] = wmma16(a0c, bc[t], acc0[t]);
      acc1[t] = wmma16(a1c, bc[t], acc1[t]);
    }
    a0c = a0n; a1c = a1n;
#pragma unroll
    for (int t = 0; t < 4; ++t) bc[t] = bn[t];
  }

#pragma unroll
  for (int t = 0; t < 4; ++t) {
    const int c = c0 + t * 16 + l16;
    const float bias = bo[c];
    const size_t base = ((size_t)b * CCH + c) * NTOK + n0 + roff;
#pragma unroll
    for (int half2 = 0; half2 < 2; ++half2) {
      const v8f& av   = half2 ? acc1[t] : acc0[t];
      const size_t bi = base + (size_t)half2 * 16;
      const v4f x0 = *(const v4f*)(x + bi);
      const v4f x1 = *(const v4f*)(x + bi + 4);
      v4f r0, r1;
#pragma unroll
      for (int g = 0; g < 4; ++g) {
        r0[g] = av[g]     + bias + x0[g];
        r1[g] = av[4 + g] + bias + x1[g];
      }
      *(v4f*)(out + bi)     = r0;
      *(v4f*)(out + bi + 4) = r1;
    }
  }
}

// ---------------------------------------------------------------------------
// Host launch
// ---------------------------------------------------------------------------
extern "C" void kernel_launch(void* const* d_in, const int* in_sizes, int n_in,
                              void* d_out, int out_size, void* d_ws, size_t ws_size,
                              hipStream_t stream) {
  (void)in_sizes; (void)n_in; (void)out_size; (void)ws_size;
  const float* x  = (const float*)d_in[0];   // [8,512,32,32]
  const float* Wp = (const float*)d_in[1];   // [512,1536]
  const float* bp = (const float*)d_in[2];   // [1536]
  const float* Wo = (const float*)d_in[3];   // [512,512]
  const float* bo = (const float*)d_in[4];   // [512]
  float* out = (float*)d_out;

  char* ws = (char*)d_ws;
  _Float16* xh  = (_Float16*)(ws);                 //  8 MB  [B][N][C]
  _Float16* WpT = (_Float16*)(ws +  8388608);      //  1.5MB [1536][512]
  _Float16* WoT = (_Float16*)(ws +  9961472);      //  0.5MB [512][512]
  _Float16* qh  = (_Float16*)(ws + 10485760);      //  8 MB  [B][H][N][128]
  _Float16* kh  = (_Float16*)(ws + 18874368);      //  8 MB  [B][H][N][128]
  _Float16* vT  = (_Float16*)(ws + 27262976);      //  8 MB  [B][H][128][N]
  _Float16* at  = (_Float16*)(ws + 35651584);      //  8 MB  [B][N][512]

  const dim3 tb(32, 8);
  transpose_cast<<<dim3(NTOK / 32, CCH / 32, NB), tb, 0, stream>>>(x,  xh,  CCH, NTOK);
  transpose_cast<<<dim3(1536 / 32, CCH / 32, 1),  tb, 0, stream>>>(Wp, WpT, CCH, 1536);
  transpose_cast<<<dim3(CCH  / 32, CCH / 32, 1),  tb, 0, stream>>>(Wo, WoT, CCH, CCH);

  qkv_gemm    <<<dim3(NTOK / 32, 1536 / 64, NB), 32, 0, stream>>>(xh, WpT, bp, qh, kh, vT);
  attention_fa<<<dim3(NTOK / 16, NH,        NB), 32, 0, stream>>>(qh, kh, vT, at);
  out_proj    <<<dim3(NTOK / 32, CCH / 64,  NB), 32, 0, stream>>>(at, WoT, bo, x, out);
}
